// AdditiveAttention_57475252355160
// MI455X (gfx1250) — matrix operand order repaired and checked by reading, not compiled, not run on hardware
//
#include <hip/hip_runtime.h>
#include <hip/hip_bf16.h>
#include <math.h>

#define B_  64
#define SK_ 2048
#define D_  1024
#define A_  1024

typedef __attribute__((ext_vector_type(16))) __bf16 v16bf;
typedef __attribute__((ext_vector_type(4)))  __bf16 v4bf;
typedef __attribute__((ext_vector_type(8)))  float  v8f;
typedef __attribute__((ext_vector_type(4)))  float  v4f;
typedef __attribute__((ext_vector_type(4)))  unsigned int v4u;
typedef __attribute__((ext_vector_type(8)))  int    v8i;
typedef __attribute__((ext_vector_type(4)))  int    v4i;

union Frag { uint4 q[2]; v16bf v; };

// keys tile: [128 rows][KSTR] bf16, resident for whole kernel
#define KSTR 1032                               // 1024 + 8 pad: conflict-free, 16B rows
#define KLDS_BYTES (128 * KSTR * 2)             // 264192
// Wk chunk: [128 a][WSTR] bf16, 64 d per chunk, double buffered
#define WSTR 72                                 // 64 + 8 pad (matches TDM pad 4DW/32DW)
#define WCH_BYTES (128 * WSTR * 2)              // 18432
#define SBUF_OFF  (KLDS_BYTES + 2 * WCH_BYTES)  // 301056
#define SMEM_BYTES (SBUF_OFF + 2 * 128 * 4)     // 302080

#if __has_builtin(__builtin_amdgcn_tensor_load_to_lds)
#define USE_TDM 1
#else
#define USE_TDM 0
#endif

typedef __attribute__((address_space(3))) unsigned char* lds_raw_t;

__device__ __forceinline__ void async_copy16(unsigned lds_addr, const void* gaddr) {
  asm volatile("global_load_async_to_lds_b128 %0, %1, off"
               :: "v"(lds_addr), "v"(gaddr) : "memory");
}
__device__ __forceinline__ void wait_async0() {
  asm volatile("s_wait_asynccnt 0x0" ::: "memory");
}

#if USE_TDM
// TDM: copy a 128(a) x 64(d) bf16 tile from WkT (row stride 1024 elems) into
// LDS at lds_byte_addr, inserting 4 DWORDs of pad after every 32 DWORDs
// (-> LDS row stride 72 elems = WSTR). One instruction, tracked by TENSORcnt.
__device__ __forceinline__ void tdm_load_chunk(const unsigned short* gsrc,
                                               unsigned lds_byte_addr) {
  const unsigned long long ga = (unsigned long long)gsrc;
  v4u g0;
  g0[0] = 1u;                                           // count=1, user mode
  g0[1] = lds_byte_addr;                                // lds_addr (bytes)
  g0[2] = (unsigned)(ga & 0xFFFFFFFFu);                 // global_addr[31:0]
  g0[3] = (unsigned)((ga >> 32) & 0x01FFFFFFu) | (2u << 30); // [56:32] | type=2
  v8i g1;
  g1[0] = (int)((1u << 16)        // data_size = 2 bytes
              | (1u << 20)        // pad_enable
              | (4u << 22)        // pad_interval: 32 DWORDs (128 B = one row)
              | (3u << 25));      // pad_amount: 4 DWORDs (16 B)
  g1[1] = (int)(1024u << 16);     // tensor_dim0 = 1024  (bits 79:48, low half)
  g1[2] = (int)(1024u << 16);     // dim0 hi = 0 | tensor_dim1 = 1024 (lo16)
  g1[3] = (int)(64u << 16);       // dim1 hi = 0 | tile_dim0 = 64
  g1[4] = (int)128u;              // tile_dim1 = 128 | tile_dim2 = 0
  g1[5] = (int)1024u;             // tensor_dim0_stride = 1024 (lo32)
  g1[6] = 0;                      // stride hi | tensor_dim1_stride lo
  g1[7] = 0;
  const v4i gz = {0, 0, 0, 0};
#if defined(__clang_major__) && (__clang_major__ >= 23)
  const v8i gz8 = {0, 0, 0, 0, 0, 0, 0, 0};
  __builtin_amdgcn_tensor_load_to_lds(g0, g1, gz, gz, gz8, 0);
#else
  __builtin_amdgcn_tensor_load_to_lds(g0, g1, gz, gz, 0);
#endif
}
#endif

__device__ __forceinline__ float fast_tanh(float x) {
#if __has_builtin(__builtin_amdgcn_tanhf)
  return __builtin_amdgcn_tanhf(x);
#elif __has_builtin(__builtin_amdgcn_tanh_f32)
  return __builtin_amdgcn_tanh_f32(x);
#else
  return tanhf(x);
#endif
}

__device__ __forceinline__ ushort4 cvt4_bf16(v4f f) {
  v4bf h = __builtin_convertvector(f, v4bf);
  return __builtin_bit_cast(ushort4, h);
}

// ---------------- q = query @ Wq : (64,1024) ----------------
__global__ void __launch_bounds__(256) qproj_kernel(
    const float* __restrict__ query, const float* __restrict__ Wq,
    float* __restrict__ qbuf) {
  __shared__ float qrow[D_];
  const int b = blockIdx.x, tid = threadIdx.x;
  for (int i = tid; i < D_; i += 256) qrow[i] = query[b * D_ + i];
  __syncthreads();
  const int a = tid * 4;
  float a0 = 0.f, a1 = 0.f, a2 = 0.f, a3 = 0.f;
  for (int d = 0; d < D_; ++d) {
    const float qd = qrow[d];
    const float4 w = *(const float4*)&Wq[(size_t)d * A_ + a];
    a0 += qd * w.x; a1 += qd * w.y; a2 += qd * w.z; a3 += qd * w.w;
  }
  float4 o; o.x = a0; o.y = a1; o.z = a2; o.w = a3;
  *(float4*)&qbuf[b * A_ + a] = o;
}

// -------- WkT[a][d] = bf16(Wk[d][a])  (one-off transpose+convert) --------
__global__ void __launch_bounds__(256) wkconv_kernel(
    const float* __restrict__ Wk, unsigned short* __restrict__ WkT) {
  __shared__ unsigned short tile[64 * 72];
  const int d0 = blockIdx.x * 64, a0 = blockIdx.y * 64, tid = threadIdx.x;
  #pragma unroll
  for (int it = 0; it < 4; ++it) {
    const int elem = it * 1024 + tid * 4;
    const int r = elem >> 6, c = elem & 63;         // r: d-local, c: a-local
    const v4f w = *(const v4f*)&Wk[(size_t)(d0 + r) * A_ + a0 + c];
    *(ushort4*)&tile[r * 72 + c] = cvt4_bf16(w);
  }
  __syncthreads();
  #pragma unroll
  for (int it = 0; it < 4; ++it) {
    const int elem = it * 1024 + tid * 4;
    const int ar = elem >> 6, dc = elem & 63;       // ar: a-local, dc: d-local
    ushort4 o;
    o.x = tile[(dc + 0) * 72 + ar];
    o.y = tile[(dc + 1) * 72 + ar];
    o.z = tile[(dc + 2) * 72 + ar];
    o.w = tile[(dc + 3) * 72 + ar];
    *(ushort4*)&WkT[(size_t)(a0 + ar) * D_ + d0 + dc] = o;
  }
}

// ------- scores[b,s] = v . tanh(q[b] + keys[b,s] @ Wk)  (WMMA bf16) -------
// grid (B, SK/128), block 512 = 16 waves. Wave pair (wave>>1) owns one s-tile;
// parity (wave&1) covers 4 of the 8 n-tiles per 128-col chunk.
// Keys (128x1024 bf16) resident in LDS; Wk B-tiles double-buffered via the
// Tensor Data Mover (TENSORcnt) or per-lane async loads (ASYNCcnt fallback).
__global__ void __launch_bounds__(512) scores_kernel(
    const float* __restrict__ keys, const unsigned short* __restrict__ WkT,
    const float* __restrict__ qbuf, const float* __restrict__ vvec,
    float* __restrict__ scores) {
  extern __shared__ unsigned char smem_raw[];
  unsigned short* klds = (unsigned short*)smem_raw;                   // [128][KSTR]
  unsigned short* wlds0 = (unsigned short*)(smem_raw + KLDS_BYTES);   // [128 a][WSTR]
  unsigned short* wlds1 = (unsigned short*)(smem_raw + KLDS_BYTES + WCH_BYTES);
  float* sbuf = (float*)(smem_raw + SBUF_OFF);                        // [2][128]
  const unsigned smem_base = (unsigned)(unsigned long long)(lds_raw_t)smem_raw;

  const int b    = blockIdx.x;
  const int s0   = blockIdx.y * 128;
  const int tid  = threadIdx.x;
  const int wave = tid >> 5;
  const int lane = tid & 31;
  const int w2   = wave >> 1;                // s-tile index 0..7
  const int p    = wave & 1;                 // n-tile parity
  const int m    = lane & 15;
  const int half = lane >> 4;
  const int koA  = half ? 8 : 0;
  const int koB  = half ? 8 : 0;

  // per-thread slice of a Wk chunk copy (async fallback): 2 x 16B
  const int el0 = tid * 8;
  const int cr0 = el0 >> 6, cc0 = el0 & 63;
  const int el1 = (512 + tid) * 8;
  const int cr1 = el1 >> 6, cc1 = el1 & 63;

  // issue copy of chunk `cc` (a0 = (cc>>4)*128, dbase = (cc&15)*64)
  auto issue_chunk = [&](int cc) {
    const int a0 = (cc >> 4) * 128;
    const int dbase = (cc & 15) * 64;
    const unsigned wbase =
        smem_base + KLDS_BYTES + (unsigned)(cc & 1) * (unsigned)WCH_BYTES;
#if USE_TDM
    tdm_load_chunk(&WkT[(size_t)a0 * D_ + dbase], wbase);
#else
    async_copy16(wbase + (unsigned)(cr0 * WSTR + cc0) * 2u,
                 &WkT[(size_t)(a0 + cr0) * D_ + dbase + cc0]);
    async_copy16(wbase + (unsigned)(cr1 * WSTR + cc1) * 2u,
                 &WkT[(size_t)(a0 + cr1) * D_ + dbase + cc1]);
#endif
  };

  if (tid < 256) sbuf[tid] = 0.f;

#if USE_TDM
  if (wave == 0) issue_chunk(0);             // overlap with keys staging
#else
  issue_chunk(0);
#endif

  // ---- stage the full keys tile once: 128 rows x 1024 d, fp32 -> bf16 ----
  #pragma unroll 4
  for (int it = 0; it < 64; ++it) {
    const int elem = it * 2048 + tid * 4;
    const int r = elem >> 10, c = elem & 1023;
    const v4f kv = *(const v4f*)&keys[((size_t)b * SK_ + s0 + r) * D_ + c];
    *(ushort4*)&klds[r * KSTR + c] = cvt4_bf16(kv);
  }

  float partial[8];
  #pragma unroll
  for (int r = 0; r < 8; ++r) partial[r] = 0.f;

  const int arow = w2 * 16 + m;

  for (int ntc = 0; ntc < 8; ++ntc) {        // a-columns [ntc*128, +128)
    v8f acc[4];
    #pragma unroll
    for (int j = 0; j < 4; ++j) {
      v8f z = {0.f, 0.f, 0.f, 0.f, 0.f, 0.f, 0.f, 0.f};
      acc[j] = z;
    }

    for (int dcc = 0; dcc < 16; ++dcc) {     // d chunk of 64
      const int cc = ntc * 16 + dcc;
      const int dbase = dcc * 64;

#if USE_TDM
      if (wave == 0) __builtin_amdgcn_s_wait_tensorcnt((short)0);
#else
      wait_async0();
#endif
      __syncthreads();                       // chunk visible; prev buf free
#if USE_TDM
      if (wave == 0 && cc + 1 < 128) issue_chunk(cc + 1);
#else
      if (cc + 1 < 128) issue_chunk(cc + 1);
#endif

      const unsigned short* wcur = (cc & 1) ? wlds1 : wlds0;

      // load all fragments first, then 8 back-to-back WMMAs
      Frag af[2], bf[2][4];
      #pragma unroll
      for (int kt = 0; kt < 2; ++kt) {
        const int kidx = dbase + kt * 32;
        af[kt].q[0] = *(const uint4*)&klds[arow * KSTR + kidx + koA];
        af[kt].q[1] = *(const uint4*)&klds[arow * KSTR + kidx + 16 + koA];
        #pragma unroll
        for (int j = 0; j < 4; ++j) {
          const int col = p * 64 + j * 16 + m;
          bf[kt][j].q[0] = *(const uint4*)&wcur[col * WSTR + kt * 32 + koB];
          bf[kt][j].q[1] = *(const uint4*)&wcur[col * WSTR + kt * 32 + koB + 16];
        }
      }
      #pragma unroll
      for (int kt = 0; kt < 2; ++kt) {
        #pragma unroll
        for (int j = 0; j < 4; ++j) {
          acc[j] = __builtin_amdgcn_wmma_f32_16x16x32_bf16(
              false, af[kt].v, false, bf[kt][j].v, (short)0, acc[j], false, false);
        }
      }
    }

    // k-tile finished for a-cols [ntc*128,+128): tanh(q+k)*v per row
    const int a0 = ntc * 128;
    #pragma unroll
    for (int j = 0; j < 4; ++j) {
      const int n  = a0 + p * 64 + j * 16 + m;
      const float qv = qbuf[b * A_ + n];
      const float vv = vvec[n];
      #pragma unroll
      for (int r = 0; r < 8; ++r)
        partial[r] += fast_tanh(acc[j][r] + qv) * vv;
    }
  }

  // reduce the 16 lanes of each half-wave (rows r / r+8)
  #pragma unroll
  for (int off = 1; off < 16; off <<= 1) {
    #pragma unroll
    for (int r = 0; r < 8; ++r)
      partial[r] += __shfl_xor(partial[r], off, 32);
  }
  if (m == 0) {
    const int row = w2 * 16 + half * 8;
    #pragma unroll
    for (int r = 0; r < 8; ++r)
      sbuf[p * 128 + row + r] = partial[r];
  }
  __syncthreads();
  if (tid < 128)
    scores[b * SK_ + s0 + tid] = sbuf[tid] + sbuf[128 + tid];
}

// ---------------- masked softmax over SK, writes attn output ----------------
__global__ void __launch_bounds__(256) softmax_kernel(
    const float* __restrict__ scores, const unsigned char* __restrict__ mask,
    float* __restrict__ attn) {
  __shared__ float red[256];
  const int b = blockIdx.x, tid = threadIdx.x;
  float loc[8];
  float mx = -INFINITY;
  #pragma unroll
  for (int i = 0; i < 8; ++i) {
    const int s = i * 256 + tid;
    float sc = scores[b * SK_ + s];
    if (!mask[b * SK_ + s]) sc = -INFINITY;
    loc[i] = sc;
    mx = fmaxf(mx, sc);
  }
  red[tid] = mx; __syncthreads();
  for (int o = 128; o > 0; o >>= 1) {
    if (tid < o) red[tid] = fmaxf(red[tid], red[tid + o]);
    __syncthreads();
  }
  mx = red[0]; __syncthreads();
  float sum = 0.f;
  #pragma unroll
  for (int i = 0; i < 8; ++i) { loc[i] = __expf(loc[i] - mx); sum += loc[i]; }
  red[tid] = sum; __syncthreads();
  for (int o = 128; o > 0; o >>= 1) {
    if (tid < o) red[tid] += red[tid + o];
    __syncthreads();
  }
  const float inv = 1.0f / red[0];
  #pragma unroll
  for (int i = 0; i < 8; ++i) attn[b * SK_ + i * 256 + tid] = loc[i] * inv;
}

// ---------------- ctx[b,e] = sum_s attn[b,s] * values[b,s,e] ----------------
__global__ void __launch_bounds__(256) ctx_kernel(
    const float* __restrict__ attn, const float* __restrict__ values,
    float* __restrict__ ctx) {
  __shared__ float at[256];
  const int b = blockIdx.x;
  const int e = blockIdx.y * 256 + threadIdx.x;
  float acc = 0.f;
  for (int sc = 0; sc < SK_; sc += 256) {
    __syncthreads();
    at[threadIdx.x] = attn[b * SK_ + sc + threadIdx.x];
    __syncthreads();
    #pragma unroll 8
    for (int i = 0; i < 256; ++i)
      acc += at[i] * values[((size_t)b * SK_ + sc + i) * (size_t)D_ + e];
  }
  ctx[b * A_ + e] = acc;
}

extern "C" void kernel_launch(void* const* d_in, const int* in_sizes, int n_in,
                              void* d_out, int out_size, void* d_ws, size_t ws_size,
                              hipStream_t stream) {
  const float*         query  = (const float*)d_in[0];
  const float*         keys   = (const float*)d_in[1];
  const float*         values = (const float*)d_in[2];
  const unsigned char* mask   = (const unsigned char*)d_in[3];
  const float*         Wq     = (const float*)d_in[4];
  const float*         Wk     = (const float*)d_in[5];
  const float*         v      = (const float*)d_in[6];

  float* qbuf   = (float*)d_ws;                       // 64*1024 f32
  float* scores = qbuf + B_ * A_;                     // 64*2048 f32
  unsigned short* WkT = (unsigned short*)((char*)d_ws + (B_ * A_ + B_ * SK_) * sizeof(float)); // 1024*1024 bf16
  float* ctx    = (float*)d_out;                      // output 0: (64,1,1024)
  float* attn   = ctx + B_ * A_;                      // output 1: (64,1,2048)

  qproj_kernel<<<dim3(B_), dim3(256), 0, stream>>>(query, Wq, qbuf);
  wkconv_kernel<<<dim3(D_ / 64, A_ / 64), dim3(256), 0, stream>>>(Wk, WkT);

  scores_kernel<<<dim3(B_, SK_ / 128), dim3(512), SMEM_BYTES, stream>>>(
      keys, WkT, qbuf, v, scores);

  softmax_kernel<<<dim3(B_), dim3(256), 0, stream>>>(scores, mask, attn);

  ctx_kernel<<<dim3(B_, A_ / 256), dim3(256), 0, stream>>>(attn, values, ctx);
}
